// SpikingRNNBase_35089882808556
// MI455X (gfx1250) — matrix operand order repaired and checked by reading, not compiled, not run on hardware
//
#include <hip/hip_runtime.h>
#include <hip/hip_bf16.h>

// ---------------------------------------------------------------------------
// Spiking LSTM (2 layers), T=1024 B=64 IN=H=512, gates 4H=2048.
//   * xg = x @ Wih^T + (bih+bhh) precomputed for all T with a WMMA f16 GEMM
//     (double-buffered LDS tiles filled by GLOBAL_LOAD_ASYNC_TO_LDS_B128).
//   * Recurrent part: persistent kernel, 16 WGs; Whh slice (128KB f16) and
//     c-state pinned in LDS for all 1024 steps; per step h (64x512 f16) is
//     async-staged into LDS, h @ Whh^T done with v_wmma_f32_16x16x32_f16,
//     Heaviside gates + c/h update, h slice published, grid barrier.
// ---------------------------------------------------------------------------

typedef __attribute__((ext_vector_type(16))) _Float16 v16h;
typedef __attribute__((ext_vector_type(8)))  float    v8f;
typedef int v4i_ __attribute__((vector_size(16)));

#define T_STEPS 1024
#define BATCH   64
#define HDIM    512
#define GDIM    2048      // 4*H
#define KDIM    512
#define NWG     16        // workgroups in recurrent kernel
#define HSLICE  32        // H columns per WG  (H/NWG)
#define WHH_LD  520       // padded LDS row stride (halves)
#define H_LD    520
#define LDS_BYTES (128*WHH_LD*2 + 64*H_LD*2 + 4*64*32*4 + 64*32*4)  // 240640

// ---------------- CDNA5 async memory->LDS helpers --------------------------

__device__ __forceinline__ void async_ld16(void* lds, const void* g) {
#if __has_builtin(__builtin_amdgcn_global_load_async_to_lds_b128)
  typedef __attribute__((address_space(1))) v4i_ gv4;
  typedef __attribute__((address_space(3))) v4i_ lv4;
  __builtin_amdgcn_global_load_async_to_lds_b128((gv4*)(void*)g, (lv4*)lds, 0, 0);
#else
  asm volatile("global_load_async_to_lds_b128 %0, %1, off"
               :: "v"((unsigned)(uintptr_t)lds), "v"(g) : "memory");
#endif
}

__device__ __forceinline__ void wait_async0() {
#if __has_builtin(__builtin_amdgcn_s_wait_asynccnt)
  __builtin_amdgcn_s_wait_asynccnt(0);
#else
  asm volatile("s_wait_asynccnt 0x0" ::: "memory");
#endif
}

// ---------------- WMMA fragment loaders (contiguous k = 16h+s, same for A and B) -----------

// A-matrix 16x32 f16: lanes 0-15 row M=L, K 0..15;
// lanes 16-31: M=L-16, K 16..31.
__device__ __forceinline__ v16h load_frag_a(const _Float16* p, int ld, int lane) {
  const int row  = lane & 15;
  const int koff = (lane & 16) ? 16 : 0;
  const _Float16* r = p + row * ld + koff;
  v16h f;
#pragma unroll
  for (int i = 0; i < 4; ++i) {
    f[2*i]   = r[2*i];
    f[2*i+1] = r[2*i+1];
    f[8+2*i] = r[8 + 2*i];
    f[9+2*i] = r[8 + 2*i + 1];
  }
  return f;
}

// B-matrix 32x16 f16 from row-major W[n][k]: lanes 0-15 -> N=lane, K=0..15;
// lanes 16-31 -> N=lane-16, K=16..31.
__device__ __forceinline__ v16h load_frag_b(const _Float16* p, int ld, int lane) {
  const int n    = lane & 15;
  const int koff = (lane & 16) ? 16 : 0;
  const _Float16* r = p + n * ld + koff;
  v16h f;
#pragma unroll
  for (int i = 0; i < 8; ++i) {
    f[2*i]   = r[2*i];
    f[2*i+1] = r[2*i+1];
  }
  return f;
}

// ---------------- helper kernels -------------------------------------------

__global__ void __launch_bounds__(256) k_f32_to_f16(const float* __restrict__ s,
                                                    _Float16* __restrict__ d, int n) {
  int i = blockIdx.x * 256 + threadIdx.x;
  if (i < n) d[i] = (_Float16)s[i];
}

__global__ void __launch_bounds__(256) k_bias_add(const float* __restrict__ a,
                                                  const float* __restrict__ b,
                                                  float* __restrict__ o, int n) {
  int i = blockIdx.x * 256 + threadIdx.x;
  if (i < n) o[i] = a[i] + b[i];
}

__global__ void __launch_bounds__(256) k_init_state(const float* __restrict__ h0l,
                                                    _Float16* __restrict__ h16,
                                                    unsigned* __restrict__ bar) {
  int i = blockIdx.x * 256 + threadIdx.x;     // 128 blocks * 256 = 32768 = B*H
  h16[i] = (_Float16)h0l[i];
  if (i == 0) { bar[0] = 0u; bar[1] = 0u; }
}

// ---------------- xg GEMM:  out[M,N] = A[M,K] @ W[N,K]^T + bias[N] ---------
// M=65536 (T*B), N=2048, K=512. Block tile 64x128, 8 waves, double-buffered
// async LDS staging, f16 in, f16 out.

__global__ void __launch_bounds__(256) k_wmma_gemm_xg(
    const _Float16* __restrict__ A,
    const _Float16* __restrict__ W,
    const float*    __restrict__ bias,
    _Float16*       __restrict__ out) {
  __shared__ _Float16 As[2][64 * 40];
  __shared__ _Float16 Bs[2][128 * 40];

  const int tid  = threadIdx.x;
  const int lane = tid & 31;
  const int w    = tid >> 5;
  const int wm   = w & 3;         // row tile (16 rows)
  const int wn   = w >> 2;        // 64-col half
  const size_t mBase = (size_t)blockIdx.y * 64;
  const int    nBase = blockIdx.x * 128;

  const int arow = tid >> 2, ak8 = (tid & 3) << 3;           // A: 1 chunk/thread
  const int brow0 = tid >> 2,         bk80 = (tid & 3) << 3; // B: chunk tid
  const int brow1 = (tid + 256) >> 2, bk81 = (tid & 3) << 3; // B: chunk tid+256

  auto issue_tile = [&](int buf, int k0) {
    async_ld16(&As[buf][arow * 40 + ak8],
               &A[(mBase + arow) * KDIM + k0 + ak8]);
    async_ld16(&Bs[buf][brow0 * 40 + bk80],
               &W[(size_t)(nBase + brow0) * KDIM + k0 + bk80]);
    async_ld16(&Bs[buf][brow1 * 40 + bk81],
               &W[(size_t)(nBase + brow1) * KDIM + k0 + bk81]);
  };

  v8f acc0 = {}, acc1 = {}, acc2 = {}, acc3 = {};

  issue_tile(0, 0);
#pragma unroll 2
  for (int ks = 0; ks < 16; ++ks) {
    const int cur = ks & 1;
    wait_async0();          // tile ks landed in LDS (only its ops outstanding)
    __syncthreads();        // visible to all waves; buf cur^1 free to refill
    if (ks + 1 < 16) issue_tile(cur ^ 1, (ks + 1) * 32);   // overlap with WMMA

    v16h af = load_frag_a(&As[cur][(wm * 16) * 40], 40, lane);
    v16h b0 = load_frag_b(&Bs[cur][(wn * 64 +  0) * 40], 40, lane);
    acc0 = __builtin_amdgcn_wmma_f32_16x16x32_f16(false, af, false, b0, (short)0, acc0, false, false);
    v16h b1 = load_frag_b(&Bs[cur][(wn * 64 + 16) * 40], 40, lane);
    acc1 = __builtin_amdgcn_wmma_f32_16x16x32_f16(false, af, false, b1, (short)0, acc1, false, false);
    v16h b2 = load_frag_b(&Bs[cur][(wn * 64 + 32) * 40], 40, lane);
    acc2 = __builtin_amdgcn_wmma_f32_16x16x32_f16(false, af, false, b2, (short)0, acc2, false, false);
    v16h b3 = load_frag_b(&Bs[cur][(wn * 64 + 48) * 40], 40, lane);
    acc3 = __builtin_amdgcn_wmma_f32_16x16x32_f16(false, af, false, b3, (short)0, acc3, false, false);
  }

  const int mloc = wm * 16 + ((lane & 16) ? 8 : 0);
  const int nl   = lane & 15;
#pragma unroll
  for (int nt = 0; nt < 4; ++nt) {
    int n = nBase + wn * 64 + nt * 16 + nl;
    float b = bias[n];
    const v8f* acc = nt == 0 ? &acc0 : nt == 1 ? &acc1 : nt == 2 ? &acc2 : &acc3;
#pragma unroll
    for (int vr = 0; vr < 8; ++vr) {
      size_t row = mBase + mloc + vr;
      out[row * GDIM + n] = (_Float16)((*acc)[vr] + b);
    }
  }
}

// ---------------- grid-wide barrier (sense-reversal, agent scope) ----------

__device__ __forceinline__ void grid_barrier(unsigned* bar) {
  __syncthreads();
  if (threadIdx.x == 0) {
    unsigned g = __hip_atomic_load(&bar[1], __ATOMIC_RELAXED, __HIP_MEMORY_SCOPE_AGENT);
    unsigned a = __hip_atomic_fetch_add(&bar[0], 1u, __ATOMIC_ACQ_REL, __HIP_MEMORY_SCOPE_AGENT);
    if (a == NWG - 1) {
      __hip_atomic_store(&bar[0], 0u, __ATOMIC_RELAXED, __HIP_MEMORY_SCOPE_AGENT);
      __hip_atomic_fetch_add(&bar[1], 1u, __ATOMIC_ACQ_REL, __HIP_MEMORY_SCOPE_AGENT);
    } else {
      while (__hip_atomic_load(&bar[1], __ATOMIC_ACQUIRE, __HIP_MEMORY_SCOPE_AGENT) == g)
        __builtin_amdgcn_s_sleep(2);
    }
  }
  __syncthreads();
}

// ---------------- persistent recurrent kernel ------------------------------

__global__ void __launch_bounds__(256) k_spiking_recur(
    const _Float16* __restrict__ xg,    // [T*B, 4H] f16
    const _Float16* __restrict__ whh,   // [4H, H]  f16
    const float*    __restrict__ c0l,   // [B, H]   f32 (initial c of this layer)
    _Float16*       __restrict__ h16,   // [B, H]   f16 state (global exchange)
    _Float16*       __restrict__ y16,   // [T*B, H] f16 out (layer0) or null
    float*          __restrict__ y32,   // [T*B, H] f32 out (layer1) or null
    float*          __restrict__ hT,    // [B, H] f32
    float*          __restrict__ cT,    // [B, H] f32
    unsigned*       __restrict__ bar,
    int is_last) {
  extern __shared__ char smem[];
  _Float16* whh_s = (_Float16*)smem;               // [4*HSLICE][WHH_LD]
  _Float16* h_s   = whh_s + 128 * WHH_LD;          // [64][H_LD]
  float*    a_s   = (float*)(h_s + 64 * H_LD);     // [4][64][32]
  float*    c_s   = a_s + 4 * 64 * 32;             // [64][32]

  const int wg   = blockIdx.x;
  const int tid  = threadIdx.x;
  const int lane = tid & 31;
  const int w    = tid >> 5;
  const int mt   = w & 3;     // 16 batch rows
  const int half = w >> 2;    // 16-col half of the 32-col slice

  // Stage this WG's Whh slice (all 4 gates) into LDS once (async DMA).
  for (int ch = tid; ch < 8192; ch += 256) {       // 128 rows * 64 chunks
    int lr = ch >> 6, k8 = (ch & 63) << 3;
    int gi = lr >> 5, c = lr & 31;
    int grow = gi * HDIM + wg * HSLICE + c;        // row in [4H, H]
    async_ld16(&whh_s[lr * WHH_LD + k8], &whh[(size_t)grow * HDIM + k8]);
  }
  for (int i = tid; i < 2048; i += 256) {          // init c slice
    int m = i >> 5, col = i & 31;
    c_s[i] = c0l[m * HDIM + wg * HSLICE + col];
  }
  wait_async0();
  __syncthreads();

  for (int t = 0; t < T_STEPS; ++t) {
    // stage full h [64 x 512] into LDS (async DMA)
    for (int ch = tid; ch < 4096; ch += 256) {
      int row = ch >> 6, k8 = (ch & 63) << 3;
      async_ld16(&h_s[row * H_LD + k8], &h16[row * HDIM + k8]);
    }
    wait_async0();
    __syncthreads();

    v8f acc0 = {}, acc1 = {}, acc2 = {}, acc3 = {};
    const _Float16* hrow = &h_s[(mt * 16) * H_LD];
#pragma unroll 4
    for (int ks = 0; ks < 16; ++ks) {
      int k0 = ks << 5;
      v16h af = load_frag_a(hrow + k0, H_LD, lane);
      v16h b0 = load_frag_b(&whh_s[(0 * 32 + half * 16) * WHH_LD + k0], WHH_LD, lane);
      acc0 = __builtin_amdgcn_wmma_f32_16x16x32_f16(false, af, false, b0, (short)0, acc0, false, false);
      v16h b1 = load_frag_b(&whh_s[(1 * 32 + half * 16) * WHH_LD + k0], WHH_LD, lane);
      acc1 = __builtin_amdgcn_wmma_f32_16x16x32_f16(false, af, false, b1, (short)0, acc1, false, false);
      v16h b2 = load_frag_b(&whh_s[(2 * 32 + half * 16) * WHH_LD + k0], WHH_LD, lane);
      acc2 = __builtin_amdgcn_wmma_f32_16x16x32_f16(false, af, false, b2, (short)0, acc2, false, false);
      v16h b3 = load_frag_b(&whh_s[(3 * 32 + half * 16) * WHH_LD + k0], WHH_LD, lane);
      acc3 = __builtin_amdgcn_wmma_f32_16x16x32_f16(false, af, false, b3, (short)0, acc3, false, false);
    }

    const int m0 = mt * 16 + ((lane & 16) ? 8 : 0);
    const int n  = (half << 4) + (lane & 15);
#pragma unroll
    for (int vr = 0; vr < 8; ++vr) {
      a_s[(0 * 64 + m0 + vr) * 32 + n] = acc0[vr];
      a_s[(1 * 64 + m0 + vr) * 32 + n] = acc1[vr];
      a_s[(2 * 64 + m0 + vr) * 32 + n] = acc2[vr];
      a_s[(3 * 64 + m0 + vr) * 32 + n] = acc3[vr];
    }
    __syncthreads();

    // elementwise: spike gates + c/h update for this slice
    for (int idx = tid; idx < 2048; idx += 256) {
      int m = idx >> 5, col = idx & 31;
      int j = (wg << 5) + col;
      const _Float16* xr = xg + ((size_t)(t * BATCH + m) << 11) + j;
      float pi = a_s[(0 * 64 + m) * 32 + col] + (float)xr[0];
      float pf = a_s[(1 * 64 + m) * 32 + col] + (float)xr[512];
      float pg = a_s[(2 * 64 + m) * 32 + col] + (float)xr[1024];
      float po = a_s[(3 * 64 + m) * 32 + col] + (float)xr[1536];
      float gi = pi >= 0.f ? 1.f : 0.f;
      float gf = pf >= 0.f ? 1.f : 0.f;
      float gg = pg >= 0.f ? 1.f : 0.f;
      float go = po >= 0.f ? 1.f : 0.f;
      float c  = c_s[idx] * gf + gi * gg;
      float h  = c * go;
      c_s[idx] = c;
      h16[(m << 9) + j] = (_Float16)h;
      size_t yi = ((size_t)(t * BATCH + m) << 9) + j;
      if (is_last) y32[yi] = h; else y16[yi] = (_Float16)h;
      if (t == T_STEPS - 1) { hT[(m << 9) + j] = h; cT[(m << 9) + j] = c; }
      if (t < T_STEPS - 1)
        __builtin_prefetch(xg + ((size_t)((t + 1) * BATCH + m) << 11) + j, 0, 1);
    }
    __threadfence();
    grid_barrier(bar);
  }
}

// ---------------------------------------------------------------------------

extern "C" void kernel_launch(void* const* d_in, const int* in_sizes, int n_in,
                              void* d_out, int out_size, void* d_ws, size_t ws_size,
                              hipStream_t stream) {
  (void)in_sizes; (void)n_in; (void)out_size; (void)ws_size;
  const float* x    = (const float*)d_in[0];
  const float* h0   = (const float*)d_in[1];   // [2,B,H]
  const float* c0   = (const float*)d_in[2];   // [2,B,H]
  const float* Wih0 = (const float*)d_in[3];
  const float* Whh0 = (const float*)d_in[4];
  const float* bih0 = (const float*)d_in[5];
  const float* bhh0 = (const float*)d_in[6];
  const float* Wih1 = (const float*)d_in[7];
  const float* Whh1 = (const float*)d_in[8];
  const float* bih1 = (const float*)d_in[9];
  const float* bhh1 = (const float*)d_in[10];
  float* out = (float*)d_out;

  const size_t M    = (size_t)T_STEPS * BATCH;          // 65536
  const size_t NX   = M * 512;                          // x / y elements
  const size_t SZ_X16 = NX * 2;                         // 64 MiB
  const size_t SZ_XG  = M * GDIM * 2;                   // 256 MiB
  const size_t SZ_W   = (size_t)GDIM * KDIM * 2;        // 2 MiB

  char* ws = (char*)d_ws;
  size_t off = 0;
  _Float16* x16     = (_Float16*)(ws + off); off += SZ_X16;
  _Float16* y016    = (_Float16*)(ws + off); off += SZ_X16;
  _Float16* xg16    = (_Float16*)(ws + off); off += SZ_XG;
  _Float16* wih0_16 = (_Float16*)(ws + off); off += SZ_W;
  _Float16* whh0_16 = (_Float16*)(ws + off); off += SZ_W;
  _Float16* wih1_16 = (_Float16*)(ws + off); off += SZ_W;
  _Float16* whh1_16 = (_Float16*)(ws + off); off += SZ_W;
  float*    b0      = (float*)(ws + off);    off += GDIM * 4;
  float*    b1      = (float*)(ws + off);    off += GDIM * 4;
  _Float16* h16     = (_Float16*)(ws + off); off += (size_t)BATCH * HDIM * 2;
  unsigned* bar     = (unsigned*)(ws + off); off += 256;

  const size_t YEL = NX;                                // 33554432
  float* h_out0 = out + YEL;
  float* h_out1 = out + YEL + BATCH * HDIM;
  float* c_out0 = out + YEL + 2 * BATCH * HDIM;
  float* c_out1 = out + YEL + 3 * BATCH * HDIM;

  // convert inputs to f16
  k_f32_to_f16<<<(unsigned)(NX / 256), 256, 0, stream>>>(x, x16, (int)NX);
  k_f32_to_f16<<<4096, 256, 0, stream>>>(Wih0, wih0_16, GDIM * KDIM);
  k_f32_to_f16<<<4096, 256, 0, stream>>>(Whh0, whh0_16, GDIM * KDIM);
  k_f32_to_f16<<<4096, 256, 0, stream>>>(Wih1, wih1_16, GDIM * KDIM);
  k_f32_to_f16<<<4096, 256, 0, stream>>>(Whh1, whh1_16, GDIM * KDIM);
  k_bias_add<<<8, 256, 0, stream>>>(bih0, bhh0, b0, GDIM);
  k_bias_add<<<8, 256, 0, stream>>>(bih1, bhh1, b1, GDIM);

  dim3 ggrid(GDIM / 128, (unsigned)(M / 64));

  // -------- layer 0 --------
  k_init_state<<<128, 256, 0, stream>>>(h0, h16, bar);
  k_wmma_gemm_xg<<<ggrid, 256, 0, stream>>>(x16, wih0_16, b0, xg16);
  k_spiking_recur<<<NWG, 256, LDS_BYTES, stream>>>(
      xg16, whh0_16, c0, h16, y016, nullptr, h_out0, c_out0, bar, 0);

  // -------- layer 1 --------
  k_init_state<<<128, 256, 0, stream>>>(h0 + BATCH * HDIM, h16, bar);
  k_wmma_gemm_xg<<<ggrid, 256, 0, stream>>>(y016, wih1_16, b1, xg16);
  k_spiking_recur<<<NWG, 256, LDS_BYTES, stream>>>(
      xg16, whh1_16, c0 + BATCH * HDIM, h16, nullptr, out, h_out1, c_out1, bar, 1);
}
